// MoE_35656818492289
// MI455X (gfx1250) — matrix operand order repaired and checked by reading, not compiled, not run on hardware
//
#include <hip/hip_runtime.h>

// ---------------------------------------------------------------------------
// MoE top-2 FFN for MI455X (gfx1250, wave32, WMMA).
// B=2, L=2048, D=1024, DFF=4096, E=8, k=2.
// Route first (gate+top2), then grouped GEMMs per expert using
// v_wmma_f32_16x16x32_bf16. M=32 tokens per workgroup; DFF processed in
// 1024-column chunks with the h-chunk staged in LDS as bf16 and immediately
// consumed by GEMM2 into persistent fp32 accumulators (fused FFN).
// fp32 weights are converted in-register via native __bf16 casts
// (v_cvt_pk_bf16_f32 path) feeding WMMA.
// ---------------------------------------------------------------------------

typedef __attribute__((ext_vector_type(16))) __bf16        v16bf;
typedef __attribute__((ext_vector_type(8)))  float         v8f;
typedef __attribute__((ext_vector_type(8)))  unsigned int  v8u;

constexpr int Bb   = 2;
constexpr int Ll   = 2048;
constexpr int Dd   = 1024;
constexpr int FF   = 4096;
constexpr int EE   = 8;
constexpr int NTOK = Bb * Ll;          // 4096 tokens
constexpr int TM   = 32;               // token tile (2 WMMA M-tiles)
constexpr int FC   = 1024;             // DFF chunk held in LDS
constexpr int MAX_TILES = NTOK / TM;

// LDS row strides in ushorts; +2 (one dword) breaks the 16-way bank conflict
// on A-fragment reads (lanes 0-15 read same k, different m).
constexpr int SA = Dd + 2;             // x tile rows
constexpr int SC = FC + 2;             // h chunk rows
constexpr size_t SMEM_BYTES = 256 + (size_t)TM * SA * 2 + (size_t)TM * SC * 2;

__device__ __forceinline__ v8f wmma_bf16(const v8u& a, const v16bf& b, v8f c) {
  return __builtin_amdgcn_wmma_f32_16x16x32_bf16(
      false, __builtin_bit_cast(v16bf, a), false, b, (short)0, c, false, false);
}

// ---------------------------------------------------------------------------
// Gate: logits = x @ Wg + bg, softmax, top-2, scatter to per-expert lists.
// One wave per token; 8 waves per block.
// ---------------------------------------------------------------------------
__global__ void __launch_bounds__(256)
moe_gate(const float* __restrict__ x, const float* __restrict__ Wg,
         const float* __restrict__ bg, int* __restrict__ counts,
         int* __restrict__ lists, float* __restrict__ wlist) {
  const int lane = threadIdx.x & 31;
  const int wave = threadIdx.x >> 5;
  const int t    = blockIdx.x * 8 + wave;

  float acc[EE];
#pragma unroll
  for (int e = 0; e < EE; ++e) acc[e] = 0.f;

  const float* xr = x + (size_t)t * Dd;
  for (int d = lane; d < Dd; d += 32) {
    const float xv = xr[d];
    const float4* wr = (const float4*)(Wg + (size_t)d * EE);
    const float4 w0 = wr[0], w1 = wr[1];
    acc[0] += xv * w0.x; acc[1] += xv * w0.y;
    acc[2] += xv * w0.z; acc[3] += xv * w0.w;
    acc[4] += xv * w1.x; acc[5] += xv * w1.y;
    acc[6] += xv * w1.z; acc[7] += xv * w1.w;
  }
#pragma unroll
  for (int e = 0; e < EE; ++e) {
#pragma unroll
    for (int off = 16; off > 0; off >>= 1)
      acc[e] += __shfl_xor(acc[e], off, 32);
  }

  if (lane == 0) {
    float lg[EE];
    float mx = -1e30f;
#pragma unroll
    for (int e = 0; e < EE; ++e) { lg[e] = acc[e] + bg[e]; mx = fmaxf(mx, lg[e]); }
    float s = 0.f;
#pragma unroll
    for (int e = 0; e < EE; ++e) { lg[e] = __expf(lg[e] - mx); s += lg[e]; }
    const float inv = 1.f / s;
    int i1 = 0; float v1 = lg[0];
#pragma unroll
    for (int e = 1; e < EE; ++e) if (lg[e] > v1) { v1 = lg[e]; i1 = e; }
    int i2 = (i1 == 0) ? 1 : 0; float v2 = lg[i2];
#pragma unroll
    for (int e = 0; e < EE; ++e)
      if (e != i1 && lg[e] > v2) { v2 = lg[e]; i2 = e; }

    int p1 = atomicAdd(&counts[i1], 1);
    lists[i1 * NTOK + p1] = t;
    wlist[i1 * NTOK + p1] = v1 * inv;
    int p2 = atomicAdd(&counts[i2], 1);
    lists[i2 * NTOK + p2] = t;
    wlist[i2 * NTOK + p2] = v2 * inv;
  }
}

// ---------------------------------------------------------------------------
// Fused grouped FFN: one block = (expert e, 32-token tile), 8 waves.
// For each DFF chunk (1024 cols):
//   GEMM1: hC = relu(xA[32xD] @ W1e[D x chunk] + b1)    -> LDS bf16
//   GEMM2: yacc += hC @ W2e[chunk x D]                  -> 128 persistent VGPRs
// Epilogue: out[tok] += w_tok * (yacc + b2) via fp32 global atomics.
// Wave w owns output columns [w*128, w*128+128) in both GEMMs; every
// B-fragment (fp32->bf16) feeds 2 WMMAs (two M-tiles).
// ---------------------------------------------------------------------------
__global__ void __launch_bounds__(256, 1)
moe_ffn(const float* __restrict__ x,  const float* __restrict__ W1,
        const float* __restrict__ b1, const float* __restrict__ W2,
        const float* __restrict__ b2, const int* __restrict__ counts,
        const int* __restrict__ lists, const float* __restrict__ wlist,
        float* __restrict__ out) {
  const int e    = blockIdx.x;
  const int tile = blockIdx.y;
  const int cnt  = counts[e];
  if (tile * TM >= cnt) return;
  const int mcnt = min(TM, cnt - tile * TM);

  extern __shared__ char smem[];
  int*            toksS = (int*)smem;                  // [32]
  float*          wlS   = (float*)(smem + 128);        // [32]
  unsigned short* xA    = (unsigned short*)(smem + 256);
  unsigned short* hC    = xA + TM * SA;

  const int tid = threadIdx.x;
  if (tid < TM) {
    int tok = 0; float w = 0.f;
    if (tid < mcnt) {
      tok = lists[e * NTOK + tile * TM + tid];
      w   = wlist[e * NTOK + tile * TM + tid];
    }
    toksS[tid] = tok;
    wlS[tid]   = w;
  }
  __syncthreads();

  // Stage x tile -> LDS bf16 (zeros for padded rows; their gate weight is 0).
  for (int idx = tid; idx < TM * Dd; idx += 256) {
    const int r = idx >> 10, c = idx & (Dd - 1);
    const float v = (r < mcnt) ? x[(size_t)toksS[r] * Dd + c] : 0.f;
    *(__bf16*)(xA + r * SA + c) = (__bf16)v;
  }
  __syncthreads();

  const int lane = tid & 31;
  const int wave = tid >> 5;
  const int hi   = lane >> 4;     // half-wave select (WMMA layout)
  const int ln   = lane & 15;
  const float* W1e = W1 + (size_t)e * Dd * FF;
  const float* W2e = W2 + (size_t)e * FF * Dd;

  v8f yacc[16] = {};              // [n-tile][m-tile] persistent C tiles

  for (int ch = 0; ch < FF / FC; ++ch) {
    // ---------- GEMM1: hC = relu(xA @ W1e[:, chunk] + b1) ----------
    for (int n = 0; n < 8; ++n) {
      const int lcol = wave * 128 + n * 16 + ln;       // col within chunk
      const int gcol = ch * FC + lcol;                 // col within DFF
      v8f c0 = {}, c1 = {};
      for (int kk = 0; kk < Dd; kk += 32) {
        v8u a0, a1; v16bf bb;
#pragma unroll
        for (int i = 0; i < 4; ++i) {                  // A: 16x32 bf16 layout
          const int k = kk + 2 * i + hi * 8;
          a0[i]     = *(const unsigned int*)(xA + ln * SA + k);
          a0[i + 4] = *(const unsigned int*)(xA + ln * SA + k + 16);
          a1[i]     = *(const unsigned int*)(xA + (16 + ln) * SA + k);
          a1[i + 4] = *(const unsigned int*)(xA + (16 + ln) * SA + k + 16);
        }
        const float* bp = W1e + (size_t)(kk + hi * 8) * FF + gcol;
#pragma unroll
        for (int i = 0; i < 8; ++i) {                  // B: fp32 -> bf16 (cvt_pk)
          bb[2 * i]     = (__bf16)bp[(size_t)(2 * i + (i & 4) * 2) * FF];
          bb[2 * i + 1] = (__bf16)bp[(size_t)(2 * i + 1 + (i & 4) * 2) * FF];
        }
        c0 = wmma_bf16(a0, bb, c0);
        c1 = wmma_bf16(a1, bb, c1);
      }
      const float bias = b1[e * FF + gcol];
#pragma unroll
      for (int r = 0; r < 8; ++r) {                    // C row M = r + hi*8
        *(__bf16*)(hC + (r + hi * 8) * SC + lcol)      = (__bf16)fmaxf(c0[r] + bias, 0.f);
        *(__bf16*)(hC + (16 + r + hi * 8) * SC + lcol) = (__bf16)fmaxf(c1[r] + bias, 0.f);
      }
    }
    __syncthreads();   // hC fully written

    // ---------- GEMM2: yacc += hC @ W2e[chunk, :] ----------
#pragma unroll
    for (int n = 0; n < 8; ++n) {                      // unrolled: const yacc idx
      const int col = wave * 128 + n * 16 + ln;
      for (int kk = 0; kk < FC; kk += 32) {
        v8u a0, a1; v16bf bb;
#pragma unroll
        for (int i = 0; i < 4; ++i) {
          const int k = kk + 2 * i + hi * 8;
          a0[i]     = *(const unsigned int*)(hC + ln * SC + k);
          a0[i + 4] = *(const unsigned int*)(hC + ln * SC + k + 16);
          a1[i]     = *(const unsigned int*)(hC + (16 + ln) * SC + k);
          a1[i + 4] = *(const unsigned int*)(hC + (16 + ln) * SC + k + 16);
        }
        const float* bp = W2e + (size_t)(ch * FC + kk + hi * 8) * Dd + col;
#pragma unroll
        for (int i = 0; i < 8; ++i) {
          bb[2 * i]     = (__bf16)bp[(size_t)(2 * i + (i & 4) * 2) * Dd];
          bb[2 * i + 1] = (__bf16)bp[(size_t)(2 * i + 1 + (i & 4) * 2) * Dd];
        }
        yacc[2 * n]     = wmma_bf16(a0, bb, yacc[2 * n]);
        yacc[2 * n + 1] = wmma_bf16(a1, bb, yacc[2 * n + 1]);
      }
    }
    __syncthreads();   // hC consumed; safe to overwrite next chunk
  }

  // ---------- Epilogue: out[tok] += w_tok * (yacc + b2) ----------
#pragma unroll
  for (int n = 0; n < 8; ++n) {
    const int col  = wave * 128 + n * 16 + ln;
    const float bias = b2[e * Dd + col];
#pragma unroll
    for (int mt = 0; mt < 2; ++mt) {
#pragma unroll
      for (int r = 0; r < 8; ++r) {
        const int row = mt * 16 + r + hi * 8;
        const float v = (yacc[2 * n + mt][r] + bias) * wlS[row]; // padded: w==0
        unsafeAtomicAdd(out + (size_t)toksS[row] * Dd + col, v);
      }
    }
  }
}

// ---------------------------------------------------------------------------
extern "C" void kernel_launch(void* const* d_in, const int* in_sizes, int n_in,
                              void* d_out, int out_size, void* d_ws, size_t ws_size,
                              hipStream_t stream) {
  const float* x  = (const float*)d_in[0];
  const float* W1 = (const float*)d_in[1];
  const float* b1 = (const float*)d_in[2];
  const float* W2 = (const float*)d_in[3];
  const float* b2 = (const float*)d_in[4];
  const float* Wg = (const float*)d_in[5];
  const float* bg = (const float*)d_in[6];
  // d_in[7] is k; reference uses k=2 (hardcoded top-2 path).

  // Workspace layout: [0,64) counts, then lists (E*NTOK int), wlist (E*NTOK f32).
  int*   counts = (int*)d_ws;
  int*   lists  = (int*)((char*)d_ws + 64);
  float* wlist  = (float*)((char*)d_ws + 64 + (size_t)EE * NTOK * sizeof(int));

  hipMemsetAsync(d_ws, 0, 64, stream);                                // counters
  hipMemsetAsync(d_out, 0, (size_t)out_size * sizeof(float), stream); // fp32 accum

  moe_gate<<<NTOK / 8, 256, 0, stream>>>(x, Wg, bg, counts, lists, wlist);

  moe_ffn<<<dim3(EE, MAX_TILES), 256, SMEM_BYTES, stream>>>(
      x, W1, b1, W2, b2, counts, lists, wlist, (float*)d_out);
}
